// PGNN_layer_42992622633782
// MI455X (gfx1250) — matrix operand order repaired and checked by reading, not compiled, not run on hardware
//
#include <hip/hip_runtime.h>

typedef __attribute__((ext_vector_type(16))) _Float16     v16h;
typedef __attribute__((ext_vector_type(8)))  float        v8f;
typedef __attribute__((ext_vector_type(4)))  unsigned int v4u;

#define NN   100000
#define KK   16
#define DIN  128
#define DOUT 128

// Padded LDS row stride for transposed W: 264 halfs = 528 B = 33*16 B.
// b128-aligned, and 528/4 = 132 dwords == 4 (mod 64 banks): the 16 lanes of a
// B-fragment load (consecutive o) hit banks 4o..4o+3 -> all 64 banks exactly
// once per half-wave: conflict-free with no swizzle. Every fragment (cb,kc) is
// then base + 16-bit immediate: cb*8448 + kc*64 (+32) <= 59616 < 65536.
#define WSTRIDE 264
#define LDS_BYTES (DOUT * WSTRIDE * 2)   // 67584 B (dynamic LDS; WGP has 320 KB)

union AFrag { v16h v; _Float16 h[16]; v4u q[2]; };
union BFrag { v16h v; v4u q[2]; };

// feature f32 -> f16 table (L2-resident gather table at half the bytes).
__global__ __launch_bounds__(256) void cvt_f16_kernel(
    const float* __restrict__ in, _Float16* __restrict__ out) {
    const long i = ((long)blockIdx.x * 256 + threadIdx.x) * 8;   // NN*DIN % 8 == 0
    const float4 f0 = *(const float4*)(in + i);
    const float4 f1 = *(const float4*)(in + i + 4);
    AFrag r;
    r.h[0] = (_Float16)f0.x; r.h[1] = (_Float16)f0.y;
    r.h[2] = (_Float16)f0.z; r.h[3] = (_Float16)f0.w;
    r.h[4] = (_Float16)f1.x; r.h[5] = (_Float16)f1.y;
    r.h[6] = (_Float16)f1.z; r.h[7] = (_Float16)f1.w;
    __builtin_nontemporal_store(r.q[0], (v4u*)(out + i));
}

template <bool USE_F16>
__global__ __launch_bounds__(256) void pgnn_wmma_kernel(
    const float*    __restrict__ feature,     // (N,128) f32
    const _Float16* __restrict__ feat16,      // (N,128) f16 (only if USE_F16)
    const float*    __restrict__ dists_max,   // (N,16)
    const int*      __restrict__ dists_argmax,// (N,16)
    const float*    __restrict__ W_hidden,    // (256,128) row-major [k][o]
    const float*    __restrict__ b_hidden,    // (128,)
    const float*    __restrict__ w_out,       // (128,)
    const float*    __restrict__ b_out,       // (1,)
    float* __restrict__ out_position,         // (N,16)
    float* __restrict__ out_structure)        // (N,128)
{
    extern __shared__ _Float16 wt[];          // wt[o*WSTRIDE + k] = W_hidden[k][o]

    const int tid = threadIdx.x;
    for (int m = tid; m < 2 * DIN * DOUT; m += 256) {
        const int o = m >> 8;       // 0..127
        const int k = m & 255;      // == tid each pass -> contiguous LDS row
        wt[o * WSTRIDE + k] = (_Float16)W_hidden[k * DOUT + o];
    }
    __syncthreads();

    const int   lane    = tid & 31;
    const int   halfSel = lane >> 4;   // 0: lanes 0-15, 1: lanes 16-31
    const int   nIdx    = lane & 15;   // A-row M / C-column N within a 16-block
    const float bout    = b_out[0];

    float bia[8], wo[8];
#pragma unroll
    for (int cb = 0; cb < 8; ++cb) {
        bia[cb] = b_hidden[cb * 16 + nIdx];
        wo[cb]  = w_out[cb * 16 + nIdx];
    }

    const int base0 = nIdx * WSTRIDE + halfSel * 8;    // B-fragment base (halfs)

    const int waveId     = (blockIdx.x << 3) + (tid >> 5);
    const int waveStride = gridDim.x << 3;

    for (int n = waveId; n < NN; n += waveStride) {
        // This lane's A-row = neighbor nIdx of node n.
        const int   gidx = dists_argmax[n * KK + nIdx];
        const float dist = dists_max[n * KK + nIdx];

        // Build 8 A fragments over the augmented K=256 input:
        // kc 0..3 -> scaled neighbor features (Wm part), kc 4..7 -> self feature (Ws part).
        // 16-bit 16x32 A layout: lane row M = nIdx; halfs[0..7] = K c0..c0+7,
        // halfs[8..15] = K c0+16..c0+23, with c0 = kc*32 + halfSel*8.
        AFrag a[8];
#pragma unroll
        for (int kc = 0; kc < 8; ++kc) {
            const bool selfPart = (kc >= 4);
            const long row      = selfPart ? (long)n : (long)gidx;
            const long col0     = (kc & 3) * 32 + halfSel * 8;
            if constexpr (USE_F16) {
                const _Float16* src = feat16 + row * DIN + col0;
                AFrag t;
                t.q[0] = *(const v4u*)(src);        // halfs 0..7
                t.q[1] = *(const v4u*)(src + 16);   // halfs 8..15
                if (selfPart) a[kc] = t;
                else          a[kc].v = t.v * (_Float16)dist;  // v_pk_mul_f16
            } else {
                const float scale = selfPart ? 1.0f : dist;
                const float* src = feature + row * DIN + col0;
                const float4 f0 = *(const float4*)(src);
                const float4 f1 = *(const float4*)(src + 4);
                const float4 f2 = *(const float4*)(src + 16);
                const float4 f3 = *(const float4*)(src + 20);
                a[kc].h[0]  = (_Float16)(f0.x * scale);
                a[kc].h[1]  = (_Float16)(f0.y * scale);
                a[kc].h[2]  = (_Float16)(f0.z * scale);
                a[kc].h[3]  = (_Float16)(f0.w * scale);
                a[kc].h[4]  = (_Float16)(f1.x * scale);
                a[kc].h[5]  = (_Float16)(f1.y * scale);
                a[kc].h[6]  = (_Float16)(f1.z * scale);
                a[kc].h[7]  = (_Float16)(f1.w * scale);
                a[kc].h[8]  = (_Float16)(f2.x * scale);
                a[kc].h[9]  = (_Float16)(f2.y * scale);
                a[kc].h[10] = (_Float16)(f2.z * scale);
                a[kc].h[11] = (_Float16)(f2.w * scale);
                a[kc].h[12] = (_Float16)(f3.x * scale);
                a[kc].h[13] = (_Float16)(f3.y * scale);
                a[kc].h[14] = (_Float16)(f3.z * scale);
                a[kc].h[15] = (_Float16)(f3.w * scale);
            }
        }

        v8f acc[8];
        const v8f zero = {};
#pragma unroll
        for (int cb = 0; cb < 8; ++cb) acc[cb] = zero;

        // One opaque base per NODE: blocks LICM of the B loads across the node
        // loop (round-1: 400+ VGPRs + scratch spills) while keeping all 128
        // ds_load_b128 as base + immediate -> per-cb clauses, partial dscnt waits.
        int base = base0;
        asm volatile("" : "+v"(base));
        const v4u* bp = (const v4u*)&wt[base];

        // 64 WMMAs: 8 column blocks x 8 K-chunks.
#pragma unroll
        for (int cb = 0; cb < 8; ++cb) {
            BFrag b[8];
#pragma unroll
            for (int kc = 0; kc < 8; ++kc) {
                b[kc].q[0] = bp[cb * 528 + kc * 4];       // +cb*8448+kc*64 B
                b[kc].q[1] = bp[cb * 528 + kc * 4 + 2];   // ... +32 B
            }
#pragma unroll
            for (int kc = 0; kc < 8; ++kc) {
                acc[cb] = __builtin_amdgcn_wmma_f32_16x16x32_f16(
                    false, a[kc].v, false, b[kc].v, (short)0, acc[cb], false, false);
            }
        }

        // Epilogue: bias + ReLU, mean over K (rows) for out_structure,
        // dot with w_out over columns for out_position.
        float pos[8];
#pragma unroll
        for (int i = 0; i < 8; ++i) pos[i] = 0.0f;

#pragma unroll
        for (int cb = 0; cb < 8; ++cb) {
            float s = 0.0f;
#pragma unroll
            for (int i = 0; i < 8; ++i) {
                // C layout: VGPR i -> row M = i + 8*halfSel, column = cb*16 + nIdx
                float h = acc[cb][i] + bia[cb];
                h = fmaxf(h, 0.0f);
                s += h;
                pos[i] += h * wo[cb];
            }
            s += __shfl_xor(s, 16, 32);           // combine rows 0-7 with 8-15
            if (halfSel == 0)
                __builtin_nontemporal_store(
                    s * (1.0f / KK), &out_structure[(long)n * DOUT + cb * 16 + nIdx]);
        }

#pragma unroll
        for (int i = 0; i < 8; ++i) {             // 16-lane butterfly within each half
            pos[i] += __shfl_xor(pos[i], 8, 32);
            pos[i] += __shfl_xor(pos[i], 4, 32);
            pos[i] += __shfl_xor(pos[i], 2, 32);
            pos[i] += __shfl_xor(pos[i], 1, 32);
        }
        if (nIdx == 0) {
#pragma unroll
            for (int i = 0; i < 8; ++i)
                __builtin_nontemporal_store(
                    pos[i] + bout, &out_position[n * KK + halfSel * 8 + i]);
        }
    }
}

extern "C" void kernel_launch(void* const* d_in, const int* in_sizes, int n_in,
                              void* d_out, int out_size, void* d_ws, size_t ws_size,
                              hipStream_t stream) {
    (void)in_sizes; (void)n_in; (void)out_size;
    const float* feature      = (const float*)d_in[0];
    const float* dists_max    = (const float*)d_in[1];
    const int*   dists_argmax = (const int*)  d_in[2];
    const float* W_hidden     = (const float*)d_in[3];
    const float* b_hidden     = (const float*)d_in[4];
    const float* w_out        = (const float*)d_in[5];
    const float* b_out        = (const float*)d_in[6];

    float* out_position  = (float*)d_out;                 // (N,16) first in tuple
    float* out_structure = (float*)d_out + (long)NN * KK; // (N,128)

    const size_t f16Bytes = (size_t)NN * DIN * sizeof(_Float16);  // 25.6 MB

    // 1250 blocks * 8 waves = 10000 waves -> exactly 10 nodes per wave.
    if (d_ws != nullptr && ws_size >= f16Bytes) {
        // f16 gather table: halves L2 gather traffic (0.8 GB -> 0.4 GB) and
        // removes all f32->f16 conversion VALU from the hot loop.
        _Float16* feat16 = (_Float16*)d_ws;
        cvt_f16_kernel<<<(NN * DIN) / (256 * 8), 256, 0, stream>>>(feature, feat16);
        pgnn_wmma_kernel<true><<<1250, 256, LDS_BYTES, stream>>>(
            feature, feat16, dists_max, dists_argmax, W_hidden, b_hidden,
            w_out, b_out, out_position, out_structure);
    } else {
        pgnn_wmma_kernel<false><<<1250, 256, LDS_BYTES, stream>>>(
            feature, nullptr, dists_max, dists_argmax, W_hidden, b_hidden,
            w_out, b_out, out_position, out_structure);
    }
}
